// Modulation_14207751815896
// MI455X (gfx1250) — matrix operand order repaired and checked by reading, not compiled, not run on hardware
//
#include <hip/hip_runtime.h>
#include <hip/hip_bf16.h>

// ---------------------------------------------------------------------------
// CDNA5 (gfx1250) fused guided-modulation kernel set.
// Precision strategy: bf16 operands + fp32 WMMA accumulation
// (v_wmma_f32_16x16x32_bf16), softmax & output in fp32.
// ---------------------------------------------------------------------------

typedef __attribute__((ext_vector_type(16))) __bf16 v16bf;
typedef __attribute__((ext_vector_type(8)))  __bf16 v8bf;
typedef __attribute__((ext_vector_type(8)))  float  v8f;
typedef __attribute__((ext_vector_type(4)))  float  v4f;

#define HH 112
#define WW 112
#define HWSZ (HH * WW)
#define CC 64

static __device__ __forceinline__ __bf16 f2bf(float f) { return (__bf16)f; }

static __device__ __forceinline__ v16bf mk16(v8bf lo, v8bf hi) {
  union { v16bf v; v8bf h[2]; } u;
  u.h[0] = lo; u.h[1] = hi;
  return u.v;
}

// A fragment (16-bit A 16x32): lanes 0-15 hold K 0..15,
// lanes 16-31 hold K 16..31 (matches b_frag). `row` = this lane's M row.
static __device__ __forceinline__ v16bf a_frag(const __bf16* row, int lane, int kf) {
  int kb = kf * 32 + ((lane & 16) ? 16 : 0);
  return mk16(*(const v8bf*)(row + kb), *(const v8bf*)(row + kb + 8));
}
// B fragment (16-bit B 32x16): lanes 0-15 hold K 0..15, lanes 16-31 K 16..31.
// `row` = this lane's N column, K-contiguous.
static __device__ __forceinline__ v16bf b_frag(const __bf16* row, int lane, int kf) {
  int kb = kf * 32 + ((lane & 16) ? 16 : 0);
  return mk16(*(const v8bf*)(row + kb), *(const v8bf*)(row + kb + 8));
}
static __device__ __forceinline__ v8f wmma_bf16(v16bf a, v16bf b, v8f c) {
  return __builtin_amdgcn_wmma_f32_16x16x32_bf16(false, a, false, b, (short)0, c,
                                                 false, false);
}

// ---------------------------------------------------------------------------
// Kernel 1: g = W1*f_guide+b1 ; s = W2*f_source+b2 (per-pixel 64x64 GEMMs).
// Output stored channel-last bf16: [b][h*W+w][c].
// 256 thr (8 waves), 128 pixels/block, one 16-pixel M-tile per wave.
// Epilogue restages D tiles through LDS so global stores are coalesced b128.
// ---------------------------------------------------------------------------
__global__ __launch_bounds__(256) void k_conv_gs(
    const float* __restrict__ fg, const float* __restrict__ fs,
    const float* __restrict__ w1, const float* __restrict__ b1,
    const float* __restrict__ w2, const float* __restrict__ b2,
    __bf16* __restrict__ g_out, __bf16* __restrict__ s_out) {
  __shared__ __bf16 xg[128][64];    // activations in, then g-result staging
  __shared__ __bf16 xs[128][64];    // activations in, then s-result staging
  __shared__ __bf16 w1b[64][64];    // weights row-major (co, ci) == B-operand layout
  __shared__ __bf16 w2b[64][64];

  const int b  = blockIdx.y;
  const int p0 = blockIdx.x * 128;
  const int t  = threadIdx.x;
  const float* fgb = fg + (size_t)b * CC * HWSZ;
  const float* fsb = fs + (size_t)b * CC * HWSZ;

  for (int i = t; i < 64 * 64; i += 256) {
    (&w1b[0][0])[i] = f2bf(w1[i]);
    (&w2b[0][0])[i] = f2bf(w2[i]);
  }
  for (int i = t; i < 128 * 64; i += 256) {
    int ci = i >> 7, pp = i & 127;   // coalesced along pixels per channel row
    xg[pp][ci] = f2bf(fgb[ci * HWSZ + p0 + pp]);
    xs[pp][ci] = f2bf(fsb[ci * HWSZ + p0 + pp]);
  }
  __syncthreads();

  const int lane = t & 31, wv = t >> 5;
  const int m = lane & 15;
  const __bf16* ag_row = &xg[wv * 16 + m][0];
  const __bf16* as_row = &xs[wv * 16 + m][0];
  v16bf ag0 = a_frag(ag_row, lane, 0), ag1 = a_frag(ag_row, lane, 1);
  v16bf as0 = a_frag(as_row, lane, 0), as1 = a_frag(as_row, lane, 1);

  v8f accg[4], accs[4];
  for (int nt = 0; nt < 4; ++nt) {
    int co = nt * 16 + m;
    v8f ag = {}, as = {};
    ag = wmma_bf16(ag0, b_frag(&w1b[co][0], lane, 0), ag);
    ag = wmma_bf16(ag1, b_frag(&w1b[co][0], lane, 1), ag);
    as = wmma_bf16(as0, b_frag(&w2b[co][0], lane, 0), as);
    as = wmma_bf16(as1, b_frag(&w2b[co][0], lane, 1), as);
    accg[nt] = ag; accs[nt] = as;
  }
  __syncthreads();  // all LDS reads complete; reuse xg/xs as output staging

  const int phl = wv * 16 + ((lane & 16) ? 8 : 0);
  for (int nt = 0; nt < 4; ++nt) {
    int co = nt * 16 + m;
    float bg = b1[co], bs = b2[co];
    for (int r = 0; r < 8; ++r) {
      xg[phl + r][co] = f2bf(accg[nt][r] + bg);
      xs[phl + r][co] = f2bf(accs[nt][r] + bs);
    }
  }
  __syncthreads();

  // coalesced wide stores, channel-last bf16
  for (int i = t; i < 128 * 8; i += 256) {
    int p = i >> 3, c8 = i & 7;
    *(v8bf*)(g_out + ((size_t)b * HWSZ + p0 + p) * 64 + c8 * 8) =
        *(const v8bf*)&xg[p][c8 * 8];
    *(v8bf*)(s_out + ((size_t)b * HWSZ + p0 + p) * 64 + c8 * 8) =
        *(const v8bf*)&xs[p][c8 * 8];
  }
}

// ---------------------------------------------------------------------------
// Kernel 2: banded neighborhood attention (11x11 and 5x5) + fusion conv.
// One block per (b, h, 16-wide w tile); 8 waves. Dynamic LDS layout below.
// ---------------------------------------------------------------------------
constexpr int OFF_SWIN  = 0;                              // [11][32][64] bf16
constexpr int OFF_SWINT = OFF_SWIN  + 11 * 32 * 64 * 2;   // [11][64][32] bf16
constexpr int OFF_GWIN  = OFF_SWINT + 11 * 64 * 32 * 2;   // [5][32][64]  bf16
constexpr int OFF_GWINT = OFF_GWIN  + 5 * 32 * 64 * 2;    // [5][64][32]  bf16
constexpr int OFF_GTILE = OFF_GWINT + 5 * 64 * 32 * 2;    // [16][64]     bf16
constexpr int OFF_SC1   = OFF_GTILE + 16 * 64 * 2;        // [16][11][32] f32
constexpr int OFF_SC2   = OFF_SC1   + 16 * 11 * 32 * 4;   // [16][5][32]  f32
constexpr int OFF_FILT1 = OFF_SC2   + 16 * 5 * 32 * 4;    // [11][16][32] bf16
constexpr int OFF_FILT2 = OFF_FILT1 + 11 * 16 * 32 * 2;   // [5][16][32]  bf16
constexpr int OFF_FEAT  = OFF_FILT2 + 5 * 16 * 32 * 2;    // [16][128]    bf16
constexpr int SMEM_BYTES = OFF_FEAT + 16 * 128 * 2;       // ~186 KB (<320 KB/WGP)

__global__ __launch_bounds__(256) void k_attn_fuse(
    const __bf16* __restrict__ g_ws, const __bf16* __restrict__ s_ws,
    const float* __restrict__ w3, const float* __restrict__ b3,
    float* __restrict__ out) {
  extern __shared__ __align__(16) char smem[];
  __bf16* s_win  = (__bf16*)(smem + OFF_SWIN);
  __bf16* s_winT = (__bf16*)(smem + OFF_SWINT);
  __bf16* g_win  = (__bf16*)(smem + OFF_GWIN);
  __bf16* g_winT = (__bf16*)(smem + OFF_GWINT);
  __bf16* g_tile = (__bf16*)(smem + OFF_GTILE);
  float*  sc1    = (float*)(smem + OFF_SC1);
  float*  sc2    = (float*)(smem + OFF_SC2);
  __bf16* filt1  = (__bf16*)(smem + OFF_FILT1);
  __bf16* filt2  = (__bf16*)(smem + OFF_FILT2);
  __bf16* feat   = (__bf16*)(smem + OFF_FEAT);

  const int t = threadIdx.x;
  const int lane = t & 31, wv = t >> 5;
  const int m = lane & 15;
  const int tile = blockIdx.x;
  const int w0 = (tile % 7) * 16;
  const int h  = (tile / 7) % HH;
  const int b  = tile / (7 * HH);
  const __bf16* gB = g_ws + (size_t)b * HWSZ * 64;
  const __bf16* sB = s_ws + (size_t)b * HWSZ * 64;

  // Issue L2/WGP prefetches (global_prefetch_b8) for the windows and W3 so
  // global latency overlaps the LDS zero-fill below.
  for (int idx = t; idx < 11 * 26; idx += 256) {
    int j = idx % 26, dy = idx / 26;
    int row = h - 5 + dy, col = w0 - 5 + j;
    if (row >= 0 && row < HH && col >= 0 && col < WW)
      __builtin_prefetch((const void*)(sB + (size_t)(row * WW + col) * 64), 0, 1);
  }
  for (int idx = t; idx < 5 * 20; idx += 256) {
    int j = idx % 20, dy = idx / 20;
    int row = h - 2 + dy, col = w0 - 2 + j;
    if (row >= 0 && row < HH && col >= 0 && col < WW)
      __builtin_prefetch((const void*)(gB + (size_t)(row * WW + col) * 64), 0, 1);
  }
  if (t < 64) __builtin_prefetch((const void*)(w3 + t * 128), 0, 1);

  { // zero LDS: pad columns / un-banded filter entries must be exactly 0
    v8bf z = {};
    __bf16* base = (__bf16*)smem;
    for (int i = t * 8; i < SMEM_BYTES / 2; i += 256 * 8) *(v8bf*)(base + i) = z;
  }
  __syncthreads();

  // Stage s window (rows h-5..h+5, cols w0-5..w0+20 -> 26, pad to 32) + transpose.
  for (int idx = t; idx < 11 * 26 * 8; idx += 256) {
    int c8 = idx & 7, j = (idx >> 3) % 26, dy = idx / (26 * 8);
    int row = h - 5 + dy, col = w0 - 5 + j;
    v8bf v = {};
    if (row >= 0 && row < HH && col >= 0 && col < WW)
      v = *(const v8bf*)(sB + (size_t)(row * WW + col) * 64 + c8 * 8);
    *(v8bf*)(s_win + (dy * 32 + j) * 64 + c8 * 8) = v;
    __bf16* tp = s_winT + (dy * 64 + c8 * 8) * 32 + j;
    for (int e = 0; e < 8; ++e) tp[e * 32] = v[e];
  }
  // Stage g window (rows h-2..h+2, cols w0-2..w0+17 -> 20, pad to 32) + transpose.
  for (int idx = t; idx < 5 * 20 * 8; idx += 256) {
    int c8 = idx & 7, j = (idx >> 3) % 20, dy = idx / (20 * 8);
    int row = h - 2 + dy, col = w0 - 2 + j;
    v8bf v = {};
    if (row >= 0 && row < HH && col >= 0 && col < WW)
      v = *(const v8bf*)(gB + (size_t)(row * WW + col) * 64 + c8 * 8);
    *(v8bf*)(g_win + (dy * 32 + j) * 64 + c8 * 8) = v;
    __bf16* tp = g_winT + (dy * 64 + c8 * 8) * 32 + j;
    for (int e = 0; e < 8; ++e) tp[e * 32] = v[e];
  }
  // Stage g center tile (16 pixels x 64 ch).
  for (int idx = t; idx < 16 * 8; idx += 256) {
    int c8 = idx & 7, p = idx >> 3;
    *(v8bf*)(g_tile + p * 64 + c8 * 8) =
        *(const v8bf*)(gB + (size_t)(h * WW + w0 + p) * 64 + c8 * 8);
  }
  __syncthreads();

  // Phase 1: correlation scores as 16x32xK=64 WMMA per window row (banded GEMM).
  // 8 waves split the 11 + 5 window rows.
  {
    v16bf ga0 = a_frag(g_tile + m * 64, lane, 0);
    v16bf ga1 = a_frag(g_tile + m * 64, lane, 1);
    const __bf16* srow = s_win + (5 * 32 + 5 + m) * 64;  // center source pixels
    v16bf sa0 = a_frag(srow, lane, 0);
    v16bf sa1 = a_frag(srow, lane, 1);
    const int phl = (lane & 16) ? 8 : 0;
    for (int dy = wv; dy < 11; dy += 8) {
      for (int jt = 0; jt < 2; ++jt) {
        const __bf16* br = s_win + (dy * 32 + jt * 16 + m) * 64;
        v8f acc = {};
        acc = wmma_bf16(ga0, b_frag(br, lane, 0), acc);
        acc = wmma_bf16(ga1, b_frag(br, lane, 1), acc);
        for (int r = 0; r < 8; ++r)
          sc1[((phl + r) * 11 + dy) * 32 + jt * 16 + m] = acc[r];
      }
    }
    for (int dy = wv; dy < 5; dy += 8) {
      for (int jt = 0; jt < 2; ++jt) {
        const __bf16* br = g_win + (dy * 32 + jt * 16 + m) * 64;
        v8f acc = {};
        acc = wmma_bf16(sa0, b_frag(br, lane, 0), acc);
        acc = wmma_bf16(sa1, b_frag(br, lane, 1), acc);
        for (int r = 0; r < 8; ++r)
          sc2[((phl + r) * 5 + dy) * 32 + jt * 16 + m] = acc[r];
      }
    }
  }
  __syncthreads();

  // Phase 2: banded softmax, fp32, one thread per pixel (band: j in [p, p+K-1]).
  if (t < 16) {
    int p = t;
    float mx = -3.0e38f, sum = 0.f;
    for (int dy = 0; dy < 11; ++dy)
      for (int dx = 0; dx < 11; ++dx)
        mx = fmaxf(mx, sc1[(p * 11 + dy) * 32 + p + dx]);
    for (int dy = 0; dy < 11; ++dy)
      for (int dx = 0; dx < 11; ++dx)
        sum += __expf(sc1[(p * 11 + dy) * 32 + p + dx] - mx);
    float inv = 1.f / sum;
    for (int dy = 0; dy < 11; ++dy)
      for (int dx = 0; dx < 11; ++dx)
        filt1[(dy * 16 + p) * 32 + p + dx] =
            f2bf(__expf(sc1[(p * 11 + dy) * 32 + p + dx] - mx) * inv);

    mx = -3.0e38f; sum = 0.f;
    for (int dy = 0; dy < 5; ++dy)
      for (int dx = 0; dx < 5; ++dx)
        mx = fmaxf(mx, sc2[(p * 5 + dy) * 32 + p + dx]);
    for (int dy = 0; dy < 5; ++dy)
      for (int dx = 0; dx < 5; ++dx)
        sum += __expf(sc2[(p * 5 + dy) * 32 + p + dx] - mx);
    inv = 1.f / sum;
    for (int dy = 0; dy < 5; ++dy)
      for (int dx = 0; dx < 5; ++dx)
        filt2[(dy * 16 + p) * 32 + p + dx] =
            f2bf(__expf(sc2[(p * 5 + dy) * 32 + p + dx] - mx) * inv);
  }
  __syncthreads();

  // Phase 3: weighted neighborhood sums (A = banded softmax weights, K=32).
  // Waves 0-3: s2g (one 16-channel tile each); waves 4-7: g2s concurrently.
  {
    const int phl = (lane & 16) ? 8 : 0;
    if (wv < 4) {
      const int c = wv * 16 + m;
      v8f acc = {};
      for (int dy = 0; dy < 11; ++dy) {
        v16bf af = a_frag(filt1 + (dy * 16 + m) * 32, lane, 0);
        v16bf bf = b_frag(s_winT + (dy * 64 + c) * 32, lane, 0);
        acc = wmma_bf16(af, bf, acc);
      }
      for (int r = 0; r < 8; ++r) feat[(phl + r) * 128 + c] = f2bf(acc[r]);
    } else {
      const int c = (wv - 4) * 16 + m;
      v8f acc = {};
      for (int dy = 0; dy < 5; ++dy) {
        v16bf af = a_frag(filt2 + (dy * 16 + m) * 32, lane, 0);
        v16bf bf = b_frag(g_winT + (dy * 64 + c) * 32, lane, 0);
        acc = wmma_bf16(af, bf, acc);
      }
      for (int r = 0; r < 8; ++r) feat[(phl + r) * 128 + 64 + c] = f2bf(acc[r]);
    }
  }
  __syncthreads();

  // Phase 4: fusion 1x1 conv, K=128, weights streamed from global (L2-resident).
  if (wv < 4) {
    const int co = wv * 16 + m;
    const int phl = (lane & 16) ? 8 : 0;
    v8f acc = {};
    for (int kf = 0; kf < 4; ++kf) {
      v16bf af = a_frag(feat + m * 128, lane, kf);
      int kb = kf * 32 + ((lane & 16) ? 16 : 0);
      const float* wr = w3 + co * 128 + kb;
      v16bf bv;
      for (int e = 0; e < 16; ++e) bv[e] = f2bf(wr[e]);
      acc = wmma_bf16(af, bv, acc);
    }
    float bias = b3[co];
    float* ob = out + ((size_t)b * CC + co) * HWSZ + h * WW + w0 + phl;
    v4f lo, hi;
    for (int r = 0; r < 4; ++r) { lo[r] = acc[r] + bias; hi[r] = acc[r + 4] + bias; }
    *(v4f*)ob = lo;
    *(v4f*)(ob + 4) = hi;
  }
}

// ---------------------------------------------------------------------------
extern "C" void kernel_launch(void* const* d_in, const int* in_sizes, int n_in,
                              void* d_out, int out_size, void* d_ws, size_t ws_size,
                              hipStream_t stream) {
  const float* f_guide  = (const float*)d_in[0];
  const float* f_source = (const float*)d_in[1];
  const float* w1 = (const float*)d_in[2];
  const float* b1 = (const float*)d_in[3];
  const float* w2 = (const float*)d_in[4];
  const float* b2 = (const float*)d_in[5];
  const float* w3 = (const float*)d_in[6];
  const float* b3 = (const float*)d_in[7];
  float* out = (float*)d_out;

  __bf16* g_ws = (__bf16*)d_ws;                       // 2*12544*64 bf16 = 3.2 MB
  __bf16* s_ws = g_ws + (size_t)2 * HWSZ * 64;        // another 3.2 MB

  (void)hipFuncSetAttribute((const void*)k_attn_fuse,
                            hipFuncAttributeMaxDynamicSharedMemorySize, SMEM_BYTES);

  dim3 g1(HWSZ / 128, 2);
  k_conv_gs<<<g1, 256, 0, stream>>>(f_guide, f_source, w1, b1, w2, b2, g_ws, s_ws);
  k_attn_fuse<<<2 * HH * 7, 256, SMEM_BYTES, stream>>>(g_ws, s_ws, w3, b3, out);
}
